// LoRALinear_8899172237474
// MI455X (gfx1250) — matrix operand order repaired and checked by reading, not compiled, not run on hardware
//
#include <hip/hip_runtime.h>
#include <hip/hip_bf16.h>

// ---------------------------------------------------------------------------
// LoRA fused linear for MI455X (gfx1250, wave32, WMMA)
//   out = x @ W + bias + scatter_e( scal * (x @ A[e]^T) @ B[e]^T )
// Base GEMM via split-bf16 (3x V_WMMA_F32_16X16X32_BF16) for ~fp32 accuracy.
// hi/lo split done once at staging time; inner loop is ds_load_b128 + WMMA.
// ---------------------------------------------------------------------------

typedef __attribute__((ext_vector_type(16))) __bf16 v16bf;
typedef __attribute__((ext_vector_type(8)))  __bf16 v8bf;
typedef __attribute__((ext_vector_type(8)))  float  v8f;
typedef int v4i __attribute__((vector_size(16)));
typedef v4i __attribute__((address_space(1)))* gptr_v4i;
typedef v4i __attribute__((address_space(3)))* lptr_v4i;

#define SEQ   8192
#define DIN   4096
#define DOUT  4096
#define RANK  16

#define BM 128
#define BN 128
#define KT 32
#define LDK 40          // padded bf16 k-stride: 80B rows, 16B-multiple
#define THREADS 512     // 16 wave32s

#if defined(__gfx1250__) && __has_builtin(__builtin_amdgcn_global_load_async_to_lds_b128)
  #define USE_ASYNC 1
#else
  #define USE_ASYNC 0
#endif

__device__ __forceinline__ void copy16_g2l(const float* g, float* l) {
#if USE_ASYNC
  // CDNA5 async global->LDS DMA (tracked by ASYNCcnt), 16B per lane.
  __builtin_amdgcn_global_load_async_to_lds_b128(
      (gptr_v4i)(const_cast<float*>(g)), (lptr_v4i)(l), 0, 0);
#else
  *reinterpret_cast<float4*>(l) = *reinterpret_cast<const float4*>(g);
#endif
}

__device__ __forceinline__ void wait_stage() {
#if USE_ASYNC
  #if __has_builtin(__builtin_amdgcn_s_wait_asynccnt)
  __builtin_amdgcn_s_wait_asynccnt(0);
  #else
  asm volatile("s_wait_asynccnt 0" ::: "memory");
  #endif
#endif
}

// ---------------------------------------------------------------------------
// Kernel 1: a_out[t][r] = scal[t] * sum_k x[t][k] * A[idx[t]][r][k]
// One block per token; x row staged to LDS via async DMA, 16 threads per rank.
// ---------------------------------------------------------------------------
__global__ __launch_bounds__(256)
void lora_a_kernel(const float* __restrict__ x, const float* __restrict__ Abuf,
                   const float* __restrict__ scal, const int* __restrict__ tok,
                   float* __restrict__ a_out) {
  __shared__ __align__(16) float xrow[DIN];
  __shared__ float red[256];
  const int t = blockIdx.x;
  const int e = tok[t];
  const float* xr = x + (long)t * DIN;
  for (int j = 0; j < 4; ++j) {
    int c = (threadIdx.x + 256 * j) << 2;          // 1024 x 16B chunks
    copy16_g2l(xr + c, &xrow[c]);
  }
  wait_stage();
  __syncthreads();
  const int r = threadIdx.x >> 4;                  // rank 0..15
  const int g = threadIdx.x & 15;                  // K-group 0..15
  const float* Ar = Abuf + ((long)e * RANK + r) * DIN;
  float acc = 0.f;
  for (int j = 0; j < DIN / 64; ++j) {             // 64 iters of float4
    int c = g + 16 * j;
    float4 xa = reinterpret_cast<const float4*>(xrow)[c];
    float4 aa = reinterpret_cast<const float4*>(Ar)[c];
    acc += xa.x * aa.x + xa.y * aa.y + xa.z * aa.z + xa.w * aa.w;
  }
  red[threadIdx.x] = acc;
  __syncthreads();
  if (g == 0) {
    float s = 0.f;
    for (int i = 0; i < 16; ++i) s += red[(r << 4) + i];
    a_out[(long)t * RANK + r] = s * scal[t];
  }
}

// ---------------------------------------------------------------------------
// Kernel 2: main GEMM (split-bf16 WMMA) + fused bias/LoRA-delta epilogue.
// 128x128 tile per WG, 16 waves in a 4x4 grid, each wave a 32x32 C strip.
// ---------------------------------------------------------------------------
__global__ __launch_bounds__(THREADS)
void lora_gemm_kernel(const float* __restrict__ x, const float* __restrict__ w,
                      const float* __restrict__ bias, const float* __restrict__ Bbuf,
                      const int* __restrict__ tok, const float* __restrict__ a_out,
                      float* __restrict__ out) {
  // bf16 tiles, hi/lo parts. A row-major [m][k]; B transposed [n][k].
  __shared__ __align__(16) __bf16 Ah[BM][LDK];
  __shared__ __align__(16) __bf16 Al[BM][LDK];
  __shared__ __align__(16) __bf16 Bh[BN][LDK];
  __shared__ __align__(16) __bf16 Bl[BN][LDK];

  const int tid  = threadIdx.x;
  const int lane = tid & 31;
  const int wave = tid >> 5;          // 0..15
  const int wm   = wave & 3;          // 32-row strip within tile
  const int wn   = wave >> 2;         // 32-col strip within tile
  const int mL   = lane & 15;
  const int hi   = lane >> 4;         // upper half-wave flag

  const long tileM = (long)blockIdx.y * BM;
  const long tileN = (long)blockIdx.x * BN;

  // Staging coordinates (hoisted out of the loop).
  const int aRow = tid >> 2;                 // 0..127
  const int aKc  = (tid & 3) << 3;           // 0,8,16,24
  const int bCol = tid & 127;                // 0..127
  const int bKs  = (tid >> 7) << 3;          // 0,8,16,24
  const float* aG = x + (tileM + aRow) * DIN + aKc;
  const float* bG = w + (long)bKs * DOUT + tileN + bCol;

  const v8f vzero = {0.f, 0.f, 0.f, 0.f, 0.f, 0.f, 0.f, 0.f};
  v8f acc[2][2];
  acc[0][0] = vzero; acc[0][1] = vzero; acc[1][0] = vzero; acc[1][1] = vzero;

  for (int k0 = 0; k0 < DIN; k0 += KT) {
    __syncthreads();                                // all waves done reading LDS
    // ---- Stage A tile: 8 contiguous f32 per thread -> bf16 hi/lo, b128 stores
    {
      const float4 f0 = reinterpret_cast<const float4*>(aG + k0)[0];
      const float4 f1 = reinterpret_cast<const float4*>(aG + k0)[1];
      const float v[8] = {f0.x, f0.y, f0.z, f0.w, f1.x, f1.y, f1.z, f1.w};
      v8bf hv, lv;
#pragma unroll
      for (int j = 0; j < 8; ++j) {
        __bf16 h = (__bf16)v[j];
        hv[j] = h;
        lv[j] = (__bf16)(v[j] - (float)h);
      }
      *reinterpret_cast<v8bf*>(&Ah[aRow][aKc]) = hv;
      *reinterpret_cast<v8bf*>(&Al[aRow][aKc]) = lv;
    }
    // ---- Stage B tile transposed: 8 k's for one n (coalesced across lanes)
    {
      const float* gp = bG + (long)k0 * DOUT;
      v8bf hv, lv;
#pragma unroll
      for (int j = 0; j < 8; ++j) {
        float vv = gp[(long)j * DOUT];
        __bf16 h = (__bf16)vv;
        hv[j] = h;
        lv[j] = (__bf16)(vv - (float)h);
      }
      *reinterpret_cast<v8bf*>(&Bh[bCol][bKs]) = hv;
      *reinterpret_cast<v8bf*>(&Bl[bCol][bKs]) = lv;
    }
    __syncthreads();

    // ---- Fragments: pure b128 LDS loads at lane-divergent bases.
    // A 16x32 bf16 layout: lanes 0-15 hold K kbA..kbA+7 / kbA+16..+23 (kbA=0),
    // lanes 16-31 the same with kbA=8.
    const int kbA = hi << 3;
    const int kbB = hi << 3;
    v16bf aHf[2], aLf[2], bHf[2], bLf[2];
#pragma unroll
    for (int tm = 0; tm < 2; ++tm) {
      const int row = wm * 32 + tm * 16 + mL;
      v8bf h0 = *reinterpret_cast<const v8bf*>(&Ah[row][kbA]);
      v8bf h1 = *reinterpret_cast<const v8bf*>(&Ah[row][kbA + 16]);
      v8bf l0 = *reinterpret_cast<const v8bf*>(&Al[row][kbA]);
      v8bf l1 = *reinterpret_cast<const v8bf*>(&Al[row][kbA + 16]);
      aHf[tm] = __builtin_shufflevector(h0, h1, 0,1,2,3,4,5,6,7,8,9,10,11,12,13,14,15);
      aLf[tm] = __builtin_shufflevector(l0, l1, 0,1,2,3,4,5,6,7,8,9,10,11,12,13,14,15);
    }
    // B 32x16: same K layout as A (lanes 0-15 K 0..7/16..23, lanes 16-31 K 8..15/24..31);
    // transposed LDS makes each 8 K-values contiguous per lane.
#pragma unroll
    for (int tn = 0; tn < 2; ++tn) {
      const int col = wn * 32 + tn * 16 + mL;
      v8bf h0 = *reinterpret_cast<const v8bf*>(&Bh[col][kbB]);
      v8bf h1 = *reinterpret_cast<const v8bf*>(&Bh[col][kbB + 16]);
      v8bf l0 = *reinterpret_cast<const v8bf*>(&Bl[col][kbB]);
      v8bf l1 = *reinterpret_cast<const v8bf*>(&Bl[col][kbB + 16]);
      bHf[tn] = __builtin_shufflevector(h0, h1, 0,1,2,3,4,5,6,7,8,9,10,11,12,13,14,15);
      bLf[tn] = __builtin_shufflevector(l0, l1, 0,1,2,3,4,5,6,7,8,9,10,11,12,13,14,15);
    }
    // ---- Split-bf16 fp32 emulation: small terms first, then dominant.
#pragma unroll
    for (int tm = 0; tm < 2; ++tm)
#pragma unroll
      for (int tn = 0; tn < 2; ++tn) {
        acc[tm][tn] = __builtin_amdgcn_wmma_f32_16x16x32_bf16(
            false, aHf[tm], false, bLf[tn], (short)0, acc[tm][tn], false, false);
        acc[tm][tn] = __builtin_amdgcn_wmma_f32_16x16x32_bf16(
            false, aLf[tm], false, bHf[tn], (short)0, acc[tm][tn], false, false);
        acc[tm][tn] = __builtin_amdgcn_wmma_f32_16x16x32_bf16(
            false, aHf[tm], false, bHf[tn], (short)0, acc[tm][tn], false, false);
      }
  }

  // Epilogue: C layout (32-bit 16x16): col = lane&15, row = vgpr + 8*(lane>=16).
  // Fuse bias and the rank-16 gathered LoRA delta (a_out row . B[e][n] row).
#pragma unroll
  for (int tm = 0; tm < 2; ++tm) {
#pragma unroll
    for (int tn = 0; tn < 2; ++tn) {
      const long n_g = tileN + wn * 32 + tn * 16 + mL;
      const float bia = bias[n_g];
#pragma unroll
      for (int r = 0; r < 8; ++r) {
        const long m_g = tileM + wm * 32 + tm * 16 + r + (hi << 3);
        const int  e   = tok[m_g];
        const float4* av = reinterpret_cast<const float4*>(a_out + m_g * RANK);
        const float4* bv = reinterpret_cast<const float4*>(Bbuf + ((long)e * DOUT + n_g) * RANK);
        float d = 0.f;
#pragma unroll
        for (int q = 0; q < 4; ++q) {
          float4 a4 = av[q];
          float4 b4 = bv[q];
          d += a4.x * b4.x + a4.y * b4.y + a4.z * b4.z + a4.w * b4.w;
        }
        out[m_g * DOUT + n_g] = acc[tm][tn][r] + bia + d;
      }
    }
  }
}

// ---------------------------------------------------------------------------
extern "C" void kernel_launch(void* const* d_in, const int* in_sizes, int n_in,
                              void* d_out, int out_size, void* d_ws, size_t ws_size,
                              hipStream_t stream) {
  (void)in_sizes; (void)n_in; (void)out_size; (void)ws_size;
  const float* x    = (const float*)d_in[0];
  const float* wgt  = (const float*)d_in[1];
  const float* bias = (const float*)d_in[2];
  const float* Ab   = (const float*)d_in[3];
  const float* Bb   = (const float*)d_in[4];
  const float* scal = (const float*)d_in[5];
  const int*   tok  = (const int*)d_in[6];
  float* out  = (float*)d_out;
  float* a_ws = (float*)d_ws;   // SEQ*RANK f32 = 512 KB scratch

  lora_a_kernel<<<SEQ, 256, 0, stream>>>(x, Ab, scal, tok, a_ws);
  dim3 grid(DOUT / BN, SEQ / BM);   // (32, 64)
  lora_gemm_kernel<<<grid, THREADS, 0, stream>>>(x, wgt, bias, Bb, tok, a_ws, out);
}
